// MultiHeadAttention_65446711656504
// MI455X (gfx1250) — matrix operand order repaired and checked by reading, not compiled, not run on hardware
//
#include <hip/hip_runtime.h>
#include <hip/hip_bf16.h>

// ---------------- types ----------------
typedef __attribute__((ext_vector_type(16))) __bf16 bf16x16;
typedef __attribute__((ext_vector_type(8)))  __bf16 bf16x8;
typedef __attribute__((ext_vector_type(4)))  __bf16 bf16x4;
typedef __attribute__((ext_vector_type(8)))  float  f32x8;
typedef __attribute__((ext_vector_type(4)))  unsigned int u32x4;
typedef __attribute__((ext_vector_type(8)))  int i32x8;
typedef __attribute__((ext_vector_type(4)))  int i32x4;

// ---------------- problem constants ----------------
constexpr int BATCH = 2;
constexpr int SEQ   = 2048;
constexpr int DM    = 1024;
constexpr int NH    = 16;
constexpr int HD    = 64;            // head dim
constexpr int MTOT  = BATCH * SEQ;   // 4096
constexpr float SCALE = 0.125f;      // 1/sqrt(64)
constexpr float NEGBIG = -3.0e38f;

static __device__ __forceinline__ f32x8 wmma_bf16(bf16x16 a, bf16x16 b, f32x8 c) {
  return __builtin_amdgcn_wmma_f32_16x16x32_bf16(
      /*neg_a=*/false, a, /*neg_b=*/false, b,
      /*c_mod=*/(short)0, c, /*reuse_a=*/false, /*reuse_b=*/false);
}

// ---------------- TDM: 2-D tile load, memory -> LDS (bf16 elements) ----------
// Builds the D# per CDNA5 ISA §8.3/8.4: group0 {count=1, lds_addr,
// global_addr(tile start), type=2}; group1 {data_size=2B, tensor_dim0/1,
// tile_dim0/1, tensor_dim0_stride}. 2-D tensor -> groups 2/3 zero.
static __device__ __forceinline__ void tdm_load_tile_2d(
    unsigned lds_off, const void* tile_gaddr,
    unsigned tensor_d0, unsigned tensor_d1,
    unsigned tile_d0, unsigned tile_d1, unsigned long long stride0) {
  unsigned long long ga = (unsigned long long)tile_gaddr;
  u32x4 g0;
  g0[0] = 1u;                                        // count=1 (valid user D#)
  g0[1] = lds_off;                                   // lds_addr (bytes)
  g0[2] = (unsigned)(ga & 0xffffffffu);              // global_addr[31:0]
  g0[3] = (unsigned)((ga >> 32) & 0x1ffffffu)        // global_addr[56:32]
        | (2u << 30);                                // type = 2 ("image")
  i32x8 g1;
  g1[0] = (1 << 16);                                 // data_size=1 -> 2 bytes
  g1[1] = (int)((tensor_d0 & 0xffffu) << 16);        // tensor_dim0[15:0]
  g1[2] = (int)(((tensor_d0 >> 16) & 0xffffu) |
                ((tensor_d1 & 0xffffu) << 16));      // td0[31:16] | td1[15:0]
  g1[3] = (int)(((tensor_d1 >> 16) & 0xffffu) |
                ((tile_d0 & 0xffffu) << 16));        // td1[31:16] | tile_dim0
  g1[4] = (int)(tile_d1 & 0xffffu);                  // tile_dim1 (tile_dim2=0)
  g1[5] = (int)(stride0 & 0xffffffffu);              // tensor_dim0_stride[31:0]
  g1[6] = (int)((stride0 >> 32) & 0xffffu);          // stride0[47:32]
  g1[7] = 0;
  i32x4 z4 = {0, 0, 0, 0};
#if defined(__clang_major__) && (__clang_major__ >= 23)
  i32x8 z8 = {0, 0, 0, 0, 0, 0, 0, 0};
  __builtin_amdgcn_tensor_load_to_lds(g0, g1, z4, z4, z8, 0);
#else
  __builtin_amdgcn_tensor_load_to_lds(g0, g1, z4, z4, 0);
#endif
}

// ---------------- f32 -> bf16 conversion ----------------
__global__ __launch_bounds__(256) void cvt_kernel(const float* __restrict__ in,
                                                  __bf16* __restrict__ out, int n4) {
  for (int i = blockIdx.x * blockDim.x + threadIdx.x; i < n4;
       i += gridDim.x * blockDim.x) {
    float4 v = ((const float4*)in)[i];
    bf16x4 o;
    o[0] = (__bf16)v.x; o[1] = (__bf16)v.y; o[2] = (__bf16)v.z; o[3] = (__bf16)v.w;
    ((bf16x4*)out)[i] = o;
  }
}

// ---------------- GEMM: C = A @ W^T + bias (TDM-staged, double-buffered) ----
// A: [M,K] bf16 row-major.  W: [N,K] bf16 row-major (torch Linear weight).
// mode 0: out bf16 in [B,H,T,Hd] layout (Q/K)
// mode 1: out bf16 in [B,H,Hd,T] layout (V transposed)
// mode 2: out f32 row-major [M,N] (final projection)
constexpr int BM = 128, BN = 64, BK = 32;
constexpr int ATILE = BM * BK;   // 4096 bf16 = 8192 B
constexpr int BTILE = BN * BK;   // 2048 bf16 = 4096 B

__global__ __launch_bounds__(256) void gemm_bf16(const __bf16* __restrict__ A,
                                                 const __bf16* __restrict__ W,
                                                 const float* __restrict__ bias,
                                                 void* __restrict__ out,
                                                 int Mdim, int Ndim, int Kdim,
                                                 int mode) {
  // single LDS object so the static byte offsets used in the D# are exact
  __shared__ __align__(128) __bf16 smem[2 * ATILE + 2 * BTILE];

  const int tid   = threadIdx.x;
  const int lane  = tid & 31;
  const int wv    = tid >> 5;        // 0..7
  const int wm    = wv >> 1;         // 0..3 (32-row strips)
  const int wn    = wv & 1;          // 0..1 (32-col strips)
  const int lrow  = lane & 15;
  const int khalf = lane >> 4;
  const int mblk  = blockIdx.y * BM;
  const int nblk  = blockIdx.x * BN;

  f32x8 acc[2][2] = {};

  // LDS byte offsets of the double buffers
  const unsigned ldsA[2] = {0u, (unsigned)(ATILE * 2)};
  const unsigned ldsB[2] = {(unsigned)(2 * ATILE * 2),
                            (unsigned)(2 * ATILE * 2 + BTILE * 2)};

  // prologue: DMA tile 0 into buffer 0 (wave 0 owns the TDM)
  if (wv == 0) {
    tdm_load_tile_2d(ldsA[0], A + (size_t)mblk * Kdim, Kdim, Mdim, BK, BM, Kdim);
    tdm_load_tile_2d(ldsB[0], W + (size_t)nblk * Kdim, Kdim, Ndim, BK, BN, Kdim);
  }

  for (int kk = 0; kk < Kdim; kk += BK) {
    const int cur = (kk / BK) & 1;
    if (wv == 0) {
      if (kk + BK < Kdim) {
        const int nxt = cur ^ 1;
        tdm_load_tile_2d(ldsA[nxt], A + (size_t)mblk * Kdim + kk + BK,
                         Kdim, Mdim, BK, BM, Kdim);
        tdm_load_tile_2d(ldsB[nxt], W + (size_t)nblk * Kdim + kk + BK,
                         Kdim, Ndim, BK, BN, Kdim);
        __builtin_amdgcn_s_wait_tensorcnt(2);   // current tile pair complete
      } else {
        __builtin_amdgcn_s_wait_tensorcnt(0);
      }
    }
    __syncthreads();                             // tiles visible to all waves

    const __bf16* As = smem + (size_t)cur * ATILE;
    const __bf16* Bs = smem + 2 * ATILE + (size_t)cur * BTILE;

    bf16x16 afrag[2], bfrag[2];
#pragma unroll
    for (int i = 0; i < 2; ++i) {
      const __bf16* ap = &As[(wm * 32 + i * 16 + lrow) * BK];
      bf16x8 lo = *(const bf16x8*)(ap + 8 * khalf);
      bf16x8 hi = *(const bf16x8*)(ap + 16 + 8 * khalf);
#pragma unroll
      for (int e = 0; e < 8; ++e) { afrag[i][e] = lo[e]; afrag[i][8 + e] = hi[e]; }
    }
#pragma unroll
    for (int j = 0; j < 2; ++j) {
      const __bf16* bp = &Bs[(wn * 32 + j * 16 + lrow) * BK];
      bf16x8 lo = *(const bf16x8*)(bp + 8 * khalf);
      bf16x8 hi = *(const bf16x8*)(bp + 16 + 8 * khalf);
#pragma unroll
      for (int e = 0; e < 8; ++e) { bfrag[j][e] = lo[e]; bfrag[j][8 + e] = hi[e]; }
    }
#pragma unroll
    for (int i = 0; i < 2; ++i)
#pragma unroll
      for (int j = 0; j < 2; ++j)
        acc[i][j] = wmma_bf16(afrag[i], bfrag[j], acc[i][j]);
    __syncthreads();   // done reading buf[cur]; next iter's DMA may overwrite it
  }

  // epilogue
#pragma unroll
  for (int i = 0; i < 2; ++i) {
#pragma unroll
    for (int j = 0; j < 2; ++j) {
#pragma unroll
      for (int r = 0; r < 8; ++r) {
        int m = mblk + wm * 32 + i * 16 + r + 8 * khalf;
        int n = nblk + wn * 32 + j * 16 + lrow;
        float v = acc[i][j][r] + bias[n];
        if (mode == 2) {
          ((float*)out)[(size_t)m * Ndim + n] = v;
        } else {
          int b = m >> 11, t = m & (SEQ - 1);      // SEQ = 2048
          int h = n >> 6,  hd = n & (HD - 1);
          __bf16 bv = (__bf16)v;
          if (mode == 0)
            ((__bf16*)out)[(((size_t)b * NH + h) * SEQ + t) * HD + hd] = bv;
          else
            ((__bf16*)out)[(((size_t)b * NH + h) * HD + hd) * SEQ + t] = bv;
        }
      }
    }
  }
}

// ---------------- fused attention: S = QK^T/sqrt(d), mask, softmax, ctx = P@V ----------------
// Grid: (B*H, SEQ/128). 8 waves; each wave owns 16 query rows.
__global__ __launch_bounds__(256) void attn_kernel(const __bf16* __restrict__ Q,
                                                   const __bf16* __restrict__ Kmat,
                                                   const __bf16* __restrict__ Vt,
                                                   const int* __restrict__ mask,
                                                   float* __restrict__ attn,
                                                   __bf16* __restrict__ ctx) {
  const int bh    = blockIdx.x;           // 0..31
  const int b     = bh / NH;
  const int h     = bh % NH;
  const int tid   = threadIdx.x;
  const int lane  = tid & 31;
  const int wv    = tid >> 5;
  const int lrow  = lane & 15;
  const int khalf = lane >> 4;
  const int qb    = blockIdx.y * 128 + wv * 16;

  const __bf16* Qbase = Q    + (size_t)bh * SEQ * HD;
  const __bf16* Kbase = Kmat + (size_t)bh * SEQ * HD;
  const __bf16* Vbase = Vt   + (size_t)bh * HD * SEQ;
  float*        Abase = attn + (size_t)bh * SEQ * SEQ;

  __shared__ float sm[8][16];
  __shared__ float sl[8][16];

  // preload Q fragments for both 32-wide K-dim steps (reused across all key tiles)
  bf16x16 qf[2];
#pragma unroll
  for (int ks = 0; ks < 2; ++ks) {
    const __bf16* qp = Qbase + (size_t)(qb + lrow) * HD + ks * 32;
    bf16x8 lo = *(const bf16x8*)(qp + 8 * khalf);
    bf16x8 hi = *(const bf16x8*)(qp + 16 + 8 * khalf);
#pragma unroll
    for (int e = 0; e < 8; ++e) { qf[ks][e] = lo[e]; qf[ks][8 + e] = hi[e]; }
  }

  float rm[8], rl[8];
#pragma unroll
  for (int r = 0; r < 8; ++r) { rm[r] = NEGBIG; rl[r] = 0.0f; }

  // ---------- phase 1: raw scores + online row stats ----------
  for (int nb = 0; nb < SEQ; nb += 64) {
    f32x8 s[4] = {};
#pragma unroll
    for (int jt = 0; jt < 4; ++jt) {
#pragma unroll
      for (int ks = 0; ks < 2; ++ks) {
        const __bf16* kp =
            Kbase + (size_t)(nb + jt * 16 + lrow) * HD + ks * 32;
        bf16x8 lo = *(const bf16x8*)(kp + 8 * khalf);
        bf16x8 hi = *(const bf16x8*)(kp + 16 + 8 * khalf);
        bf16x16 kf;
#pragma unroll
        for (int e = 0; e < 8; ++e) { kf[e] = lo[e]; kf[8 + e] = hi[e]; }
        s[jt] = wmma_bf16(qf[ks], kf, s[jt]);
      }
    }
    // scale + key-padding mask (column = this lane's key)
#pragma unroll
    for (int jt = 0; jt < 4; ++jt) {
      int key = nb + jt * 16 + lrow;
      int mv  = mask[(size_t)b * SEQ + key];
#pragma unroll
      for (int r = 0; r < 8; ++r) {
        float v = s[jt][r] * SCALE;
        s[jt][r] = mv ? v : NEGBIG;
      }
    }
    // per-row online max / sum-exp + stream raw scores out (coalesced per row)
#pragma unroll
    for (int r = 0; r < 8; ++r) {
      float mx = fmaxf(fmaxf(s[0][r], s[1][r]), fmaxf(s[2][r], s[3][r]));
#pragma unroll
      for (int off = 8; off >= 1; off >>= 1) mx = fmaxf(mx, __shfl_xor(mx, off, 32));
      float mnew = fmaxf(rm[r], mx);
      float e = __expf(s[0][r] - mnew) + __expf(s[1][r] - mnew) +
                __expf(s[2][r] - mnew) + __expf(s[3][r] - mnew);
#pragma unroll
      for (int off = 8; off >= 1; off >>= 1) e += __shfl_xor(e, off, 32);
      rl[r] = rl[r] * __expf(rm[r] - mnew) + e;
      rm[r] = mnew;

      float* dst = Abase + (size_t)(qb + r + 8 * khalf) * SEQ + nb;
#pragma unroll
      for (int jt = 0; jt < 4; ++jt) dst[jt * 16 + lrow] = s[jt][r];
    }
  }

  // broadcast per-row (m, l) across the wave via LDS
  if (lrow == 0) {
#pragma unroll
    for (int r = 0; r < 8; ++r) {
      sm[wv][r + 8 * khalf] = rm[r];
      sl[wv][r + 8 * khalf] = rl[r];
    }
  }
  __syncthreads();
  const float mrow = sm[wv][lrow];
  const float invl = 1.0f / sl[wv][lrow];

  // make phase-1 global stores visible before re-reading them
  asm volatile("s_wait_storecnt 0x0" ::: "memory");

  // ---------- phase 2: normalize P, write softmax out, ctx = P @ V ----------
  f32x8 c[4] = {};
  for (int kb = 0; kb < SEQ; kb += 32) {
    float* sp = Abase + (size_t)(qb + lrow) * SEQ + kb;
    // vectorized re-read of raw scores (16B aligned)
    float4 r0 = ((const float4*)(sp + 8 * khalf))[0];
    float4 r1 = ((const float4*)(sp + 8 * khalf))[1];
    float4 r2 = ((const float4*)(sp + 16 + 8 * khalf))[0];
    float4 r3 = ((const float4*)(sp + 16 + 8 * khalf))[1];
    float pv[16] = {r0.x, r0.y, r0.z, r0.w, r1.x, r1.y, r1.z, r1.w,
                    r2.x, r2.y, r2.z, r2.w, r3.x, r3.y, r3.z, r3.w};
    bf16x16 pf;
#pragma unroll
    for (int e = 0; e < 16; ++e) {
      float v = __expf(pv[e] - mrow) * invl;
      pv[e] = v;
      pf[e] = (__bf16)v;
    }
    // vectorized write-back of normalized softmax to d_out
    ((float4*)(sp + 8 * khalf))[0]      = make_float4(pv[0], pv[1], pv[2], pv[3]);
    ((float4*)(sp + 8 * khalf))[1]      = make_float4(pv[4], pv[5], pv[6], pv[7]);
    ((float4*)(sp + 16 + 8 * khalf))[0] = make_float4(pv[8], pv[9], pv[10], pv[11]);
    ((float4*)(sp + 16 + 8 * khalf))[1] = make_float4(pv[12], pv[13], pv[14], pv[15]);

#pragma unroll
    for (int jt = 0; jt < 4; ++jt) {
      const __bf16* vp = Vbase + (size_t)(jt * 16 + lrow) * SEQ + kb;
      bf16x8 lo = *(const bf16x8*)(vp + 8 * khalf);
      bf16x8 hi = *(const bf16x8*)(vp + 16 + 8 * khalf);
      bf16x16 vf;
#pragma unroll
      for (int e = 0; e < 8; ++e) { vf[e] = lo[e]; vf[8 + e] = hi[e]; }
      c[jt] = wmma_bf16(pf, vf, c[jt]);
    }
  }

  // ctx epilogue: [B,T,D] bf16, col = h*64 + jt*16 + lrow
#pragma unroll
  for (int jt = 0; jt < 4; ++jt) {
#pragma unroll
    for (int r = 0; r < 8; ++r) {
      int q = qb + r + 8 * khalf;
      ctx[((size_t)(b * SEQ + q)) * DM + h * HD + jt * 16 + lrow] = (__bf16)c[jt][r];
    }
  }
}

// ---------------- host launch ----------------
extern "C" void kernel_launch(void* const* d_in, const int* in_sizes, int n_in,
                              void* d_out, int out_size, void* d_ws, size_t ws_size,
                              hipStream_t stream) {
  const float* x    = (const float*)d_in[0];
  const int*   mask = (const int*)d_in[1];
  const float* Wq   = (const float*)d_in[2];
  const float* bq   = (const float*)d_in[3];
  const float* Wk   = (const float*)d_in[4];
  const float* bk   = (const float*)d_in[5];
  const float* Wv   = (const float*)d_in[6];
  const float* bv   = (const float*)d_in[7];
  const float* Wo   = (const float*)d_in[8];
  const float* bo   = (const float*)d_in[9];

  float* out_proj = (float*)d_out;                       // [B,T,D]
  float* out_attn = (float*)d_out + (size_t)MTOT * DM;   // [B,H,T,T]

  // workspace carve-up (bf16 buffers), 256B aligned
  char* ws = (char*)d_ws;
  auto align256 = [](size_t v) { return (v + 255) & ~(size_t)255; };
  size_t off = 0;
  __bf16* xb  = (__bf16*)(ws + off); off = align256(off + (size_t)MTOT * DM * 2);
  __bf16* wqb = (__bf16*)(ws + off); off = align256(off + (size_t)DM * DM * 2);
  __bf16* wkb = (__bf16*)(ws + off); off = align256(off + (size_t)DM * DM * 2);
  __bf16* wvb = (__bf16*)(ws + off); off = align256(off + (size_t)DM * DM * 2);
  __bf16* wob = (__bf16*)(ws + off); off = align256(off + (size_t)DM * DM * 2);
  __bf16* Qb  = (__bf16*)(ws + off); off = align256(off + (size_t)MTOT * DM * 2);
  __bf16* Kb  = (__bf16*)(ws + off); off = align256(off + (size_t)MTOT * DM * 2);
  __bf16* Vtb = (__bf16*)(ws + off); off = align256(off + (size_t)MTOT * DM * 2);
  __bf16* ctxb= (__bf16*)(ws + off); off = align256(off + (size_t)MTOT * DM * 2);

  // f32 -> bf16
  cvt_kernel<<<2048, 256, 0, stream>>>(x,  xb,  MTOT * DM / 4);
  cvt_kernel<<<1024, 256, 0, stream>>>(Wq, wqb, DM * DM / 4);
  cvt_kernel<<<1024, 256, 0, stream>>>(Wk, wkb, DM * DM / 4);
  cvt_kernel<<<1024, 256, 0, stream>>>(Wv, wvb, DM * DM / 4);
  cvt_kernel<<<1024, 256, 0, stream>>>(Wo, wob, DM * DM / 4);

  dim3 ggrid(DM / BN, MTOT / BM);   // (16, 32)
  gemm_bf16<<<ggrid, 256, 0, stream>>>(xb, wqb, bq, Qb,  MTOT, DM, DM, 0);
  gemm_bf16<<<ggrid, 256, 0, stream>>>(xb, wkb, bk, Kb,  MTOT, DM, DM, 0);
  gemm_bf16<<<ggrid, 256, 0, stream>>>(xb, wvb, bv, Vtb, MTOT, DM, DM, 1);

  dim3 agrid(BATCH * NH, SEQ / 128);  // (32, 16)
  attn_kernel<<<agrid, 256, 0, stream>>>(Qb, Kb, Vtb, mask, out_attn, ctxb);

  gemm_bf16<<<ggrid, 256, 0, stream>>>(ctxb, wob, bo, out_proj, MTOT, DM, DM, 2);
}
